// MoELayer_85564338471392
// MI455X (gfx1250) — matrix operand order repaired and checked by reading, not compiled, not run on hardware
//
#include <hip/hip_runtime.h>
#include <cstdint>

// ---------------------------------------------------------------------------
// MoE top-2 FFN for MI455X (gfx1250, wave32):
//   router -> padded per-expert dispatch -> bf16 WMMA GEMMs (fp32 accum)
//   with async global->LDS double-buffered tile staging (ASYNCcnt).
// Workspace requirement: ~246 MB.
// ---------------------------------------------------------------------------

#define HIDDEN   1024
#define INNER    2048
#define NEXP     8
#define NTOK     8192            // B*S
#define NASSIGN  (NTOK * 2)      // top-2
#define MT       128             // GEMM M-tile
#define ROWS_CAP (NASSIGN + NEXP * MT)   // 17408
#define NTILES_M (ROWS_CAP / MT)         // 136

typedef __attribute__((ext_vector_type(16))) __bf16 v16bf;
typedef __attribute__((ext_vector_type(8)))  float  v8f;
typedef int v4i __attribute__((vector_size(16)));    // matches builtin proto

union FragBits { uint32_t u[8]; v16bf v; };

#if defined(__AMDGCN__) && __has_builtin(__builtin_amdgcn_global_load_async_to_lds_b128)
#define MOE_ASYNC 1
#endif

#define AS1 __attribute__((address_space(1)))
#define AS3 __attribute__((address_space(3)))

#ifdef MOE_ASYNC
#define ASYNC_WAIT() asm volatile("s_wait_asynccnt 0" ::: "memory")
#else
#define ASYNC_WAIT()
#endif

__device__ __forceinline__ uint16_t f2bf(float f) {
  uint32_t u = __float_as_uint(f);
  uint32_t r = u + 0x7FFFu + ((u >> 16) & 1u);   // RNE fp32 -> bf16
  return (uint16_t)(r >> 16);
}

// copy one 16B chunk global(bf16) -> LDS, async if available
__device__ __forceinline__ void stage16(const uint16_t* gsrc, uint16_t* ldst) {
#ifdef MOE_ASYNC
  __builtin_amdgcn_global_load_async_to_lds_b128((AS1 v4i*)gsrc, (AS3 v4i*)ldst,
                                                 0, 0);
#else
  *(uint4*)ldst = *(const uint4*)gsrc;
#endif
}

// ---------------------------------------------------------------- init ------
__global__ void k_init(int* counts, int* row_token) {
  int i = blockIdx.x * blockDim.x + threadIdx.x;
  if (i < NEXP) counts[i] = 0;
  if (i < ROWS_CAP) row_token[i] = -1;
}

// -------------------------------------------------------------- router ------
__global__ void k_router(const float* __restrict__ x,
                         const float* __restrict__ Wr,
                         const float* __restrict__ br,
                         int* __restrict__ counts,
                         int* __restrict__ expert_a,
                         int* __restrict__ slot_a,
                         float* __restrict__ weight_a) {
  int t = blockIdx.x * blockDim.x + threadIdx.x;
  if (t >= NTOK) return;
  const float* xr = x + (size_t)t * HIDDEN;
  float acc[NEXP];
#pragma unroll
  for (int e = 0; e < NEXP; ++e) acc[e] = br[e];
  for (int h = 0; h < HIDDEN; ++h) {
    float xv = xr[h];
    const float* wrow = Wr + h * NEXP;
#pragma unroll
    for (int e = 0; e < NEXP; ++e) acc[e] = fmaf(xv, wrow[e], acc[e]);
  }
  int e0 = 0; float l0 = acc[0];
#pragma unroll
  for (int e = 1; e < NEXP; ++e) if (acc[e] > l0) { l0 = acc[e]; e0 = e; }
  int e1 = -1; float l1 = -3.4e38f;
#pragma unroll
  for (int e = 0; e < NEXP; ++e) if (e != e0 && acc[e] > l1) { l1 = acc[e]; e1 = e; }
  float z  = expf(l1 - l0);
  float w0 = 1.0f / (1.0f + z);
  float w1 = z / (1.0f + z);
  int s0 = atomicAdd(&counts[e0], 1);
  int s1 = atomicAdd(&counts[e1], 1);
  expert_a[2 * t]     = e0; slot_a[2 * t]     = s0; weight_a[2 * t]     = w0;
  expert_a[2 * t + 1] = e1; slot_a[2 * t + 1] = s1; weight_a[2 * t + 1] = w1;
}

// ---------------------------------------------------------------- scan ------
__global__ void k_scan(const int* __restrict__ counts, int* __restrict__ poffsets) {
  if (blockIdx.x == 0 && threadIdx.x == 0) {
    int off = 0;
    for (int e = 0; e < NEXP; ++e) {
      poffsets[e] = off;
      off += ((counts[e] + MT - 1) / MT) * MT;
    }
    poffsets[NEXP] = off;
  }
}

// ------------------------------------------------------------- scatter ------
__global__ void k_scatter(const int* __restrict__ expert_a,
                          const int* __restrict__ slot_a,
                          const int* __restrict__ poffsets,
                          int* __restrict__ row_token,
                          int* __restrict__ assign_row) {
  int a = blockIdx.x * blockDim.x + threadIdx.x;
  if (a >= NASSIGN) return;
  int e   = expert_a[a];
  int row = poffsets[e] + slot_a[a];
  row_token[row] = a >> 1;
  assign_row[a]  = row;
}

// ------------------------------------------------- gather x -> Xg bf16 ------
__global__ void k_gather_x(const float* __restrict__ x,
                           const int* __restrict__ row_token,
                           uint16_t* __restrict__ Xg) {
  int idx = blockIdx.x * blockDim.x + threadIdx.x;   // ROWS_CAP * HIDDEN/8
  int r = idx >> 7;                                  // HIDDEN/8 = 128
  int c = (idx & 127) * 8;
  if (r >= ROWS_CAP) return;
  int tok = row_token[r];
  union { uint16_t h[8]; uint4 u; } o;
  if (tok < 0) {
    o.u = uint4{0, 0, 0, 0};
  } else {
    const float* src = x + (size_t)tok * HIDDEN + c;
    float4 a = *(const float4*)src;
    float4 b = *(const float4*)(src + 4);
    o.h[0] = f2bf(a.x); o.h[1] = f2bf(a.y); o.h[2] = f2bf(a.z); o.h[3] = f2bf(a.w);
    o.h[4] = f2bf(b.x); o.h[5] = f2bf(b.y); o.h[6] = f2bf(b.z); o.h[7] = f2bf(b.w);
  }
  *(uint4*)(Xg + (size_t)r * HIDDEN + c) = o.u;
}

// ---------------------- W [E][K][N] fp32 -> Wt [E][N][K] bf16 (tiled) -------
__global__ __launch_bounds__(256)
void k_transpose_w(const float* __restrict__ W, uint16_t* __restrict__ Wt,
                   int K, int N) {
  __shared__ float tile[32][33];
  int e  = blockIdx.z;
  int nb = blockIdx.x * 32, kb = blockIdx.y * 32;
  int tx = threadIdx.x & 31, ty = threadIdx.x >> 5;   // ty 0..7
  const float* Wp  = W  + (size_t)e * K * N;
  uint16_t*    Wtp = Wt + (size_t)e * N * K;
#pragma unroll
  for (int j = 0; j < 4; ++j)
    tile[ty + 8 * j][tx] = Wp[(size_t)(kb + ty + 8 * j) * N + nb + tx];
  __syncthreads();
#pragma unroll
  for (int j = 0; j < 4; ++j)
    Wtp[(size_t)(nb + ty + 8 * j) * K + kb + tx] = f2bf(tile[tx][ty + 8 * j]);
}

// ------------------------------------------------------ WMMA GEMM core ------
// C[R..R+127][nBase..nBase+127] = A[rows][KDIM] * B[e][NCOLS][KDIM]^T
// A,B bf16 row(K)-contiguous; LDS double-buffered via async global->LDS.
// Per k-step a tile is 128 rows x 32 halfs = 512 x 16B chunks; 256 threads
// stage 2 chunks each per tile.
template <int KDIM, int NCOLS, bool RELU>
__global__ __launch_bounds__(256)
void k_moe_gemm(const uint16_t* __restrict__ Abf,     // [>=rows][KDIM]
                const uint16_t* __restrict__ Bbf,     // [NEXP][NCOLS][KDIM]
                const float*    __restrict__ biasAll, // [NEXP][NCOLS]
                const int*      __restrict__ poffsets,
                uint16_t*       __restrict__ HoutBf,  // RELU=1: bf16 out
                float*          __restrict__ YoutF) { // RELU=0: fp32 out
  const int R = blockIdx.y * MT;
  if (R >= poffsets[NEXP]) return;                    // uniform per block
  int e = 0;
#pragma unroll
  for (int i = 1; i < NEXP; ++i) if (R >= poffsets[i]) e = i;

  const int nBase = blockIdx.x * 128;

  __shared__ uint16_t As[2][MT][40];
  __shared__ uint16_t Bs[2][128][40];

  const int  tid   = threadIdx.x;
  const int  lane  = tid & 31;
  const int  wave  = tid >> 5;
  const int  waveM = wave & 3;
  const int  waveN = wave >> 2;
  const int  lm    = lane & 15;
  const bool hi    = lane >= 16;

  // staging chunk coordinates (2 x 16B chunks per thread per tile)
  const int cr0 = tid >> 2,         cs0 = (tid & 3) * 8;
  const int cr1 = (tid + 256) >> 2, cs1 = ((tid + 256) & 3) * 8;

  const uint16_t* Atile = Abf + (size_t)R * KDIM;
  const uint16_t* Btile = Bbf + ((size_t)e * NCOLS + nBase) * KDIM;

  v8f acc[2][4];
#pragma unroll
  for (int i = 0; i < 2; ++i)
#pragma unroll
    for (int j = 0; j < 4; ++j) acc[i][j] = v8f{0, 0, 0, 0, 0, 0, 0, 0};

  constexpr int NS = KDIM / 32;

  // preload tile 0
  stage16(Atile + (size_t)cr0 * KDIM + cs0, &As[0][cr0][cs0]);
  stage16(Atile + (size_t)cr1 * KDIM + cs1, &As[0][cr1][cs1]);
  stage16(Btile + (size_t)cr0 * KDIM + cs0, &Bs[0][cr0][cs0]);
  stage16(Btile + (size_t)cr1 * KDIM + cs1, &Bs[0][cr1][cs1]);
  ASYNC_WAIT();
  __syncthreads();

  for (int i = 0; i < NS; ++i) {
    const int cur = i & 1;
    if (i + 1 < NS) {                       // stage next tile into other buffer
      const int k1 = (i + 1) * 32;
      stage16(Atile + (size_t)cr0 * KDIM + k1 + cs0, &As[1 - cur][cr0][cs0]);
      stage16(Atile + (size_t)cr1 * KDIM + k1 + cs1, &As[1 - cur][cr1][cs1]);
      stage16(Btile + (size_t)cr0 * KDIM + k1 + cs0, &Bs[1 - cur][cr0][cs0]);
      stage16(Btile + (size_t)cr1 * KDIM + k1 + cs1, &Bs[1 - cur][cr1][cs1]);
    }

    // fragments from current buffer (ISA 7.12.2 16-bit layouts)
    v16bf afrag[2], bfrag[4];
#pragma unroll
    for (int sm = 0; sm < 2; ++sm) {
      int m = waveM * 32 + sm * 16 + lm;
      FragBits fb;
#pragma unroll
      for (int p = 0; p < 8; ++p) {
        int kk = 2 * p + (p < 4 ? 0 : 8) + (hi ? 8 : 0);
        fb.u[p] = *(const uint32_t*)&As[cur][m][kk];
      }
      afrag[sm] = fb.v;
    }
#pragma unroll
    for (int sn = 0; sn < 4; ++sn) {
      int n = waveN * 64 + sn * 16 + lm;
      FragBits fb;
#pragma unroll
      for (int p = 0; p < 8; ++p) {
        int kk = 2 * p + (p < 4 ? 0 : 8) + (hi ? 8 : 0);
        fb.u[p] = *(const uint32_t*)&Bs[cur][n][kk];
      }
      bfrag[sn] = fb.v;
    }

#pragma unroll
    for (int sm = 0; sm < 2; ++sm)
#pragma unroll
      for (int sn = 0; sn < 4; ++sn)
        acc[sm][sn] = __builtin_amdgcn_wmma_f32_16x16x32_bf16(
            false, afrag[sm], false, bfrag[sn], (short)0, acc[sm][sn],
            false, false);

    ASYNC_WAIT();        // next tile landed (no-op on fallback path)
    __syncthreads();     // everyone done reading cur; next buffer visible
  }

  const float* bias = biasAll + (size_t)e * NCOLS;
#pragma unroll
  for (int sn = 0; sn < 4; ++sn) {
    int   col = nBase + waveN * 64 + sn * 16 + lm;
    float bv  = bias[col];
#pragma unroll
    for (int sm = 0; sm < 2; ++sm) {
#pragma unroll
      for (int r = 0; r < 8; ++r) {
        int   row = R + waveM * 32 + sm * 16 + r + (hi ? 8 : 0);
        float v   = acc[sm][sn][r] + bv;
        if (RELU) {
          v = v > 0.0f ? v : 0.0f;
          HoutBf[(size_t)row * NCOLS + col] = f2bf(v);
        } else {
          YoutF[(size_t)row * NCOLS + col] = v;
        }
      }
    }
  }
}

// ------------------------------------------------------------- combine ------
__global__ void k_combine(const float* __restrict__ Yg,
                          const int*   __restrict__ assign_row,
                          const float* __restrict__ weight_a,
                          float*       __restrict__ out) {
  int idx = blockIdx.x * blockDim.x + threadIdx.x;
  int t   = idx / (HIDDEN / 4);
  int c   = (idx % (HIDDEN / 4)) * 4;
  if (t >= NTOK) return;
  int   r0 = assign_row[2 * t],   r1 = assign_row[2 * t + 1];
  float w0 = weight_a[2 * t],     w1 = weight_a[2 * t + 1];
  float4 y0 = *(const float4*)(Yg + (size_t)r0 * HIDDEN + c);
  float4 y1 = *(const float4*)(Yg + (size_t)r1 * HIDDEN + c);
  float4 o;
  o.x = w0 * y0.x + w1 * y1.x;
  o.y = w0 * y0.y + w1 * y1.y;
  o.z = w0 * y0.z + w1 * y1.z;
  o.w = w0 * y0.w + w1 * y1.w;
  *(float4*)(out + (size_t)t * HIDDEN + c) = o;
}

// ------------------------------------------------------------- launcher -----
extern "C" void kernel_launch(void* const* d_in, const int* in_sizes, int n_in,
                              void* d_out, int out_size, void* d_ws, size_t ws_size,
                              hipStream_t stream) {
  (void)in_sizes; (void)n_in; (void)out_size; (void)ws_size;
  const float* x  = (const float*)d_in[0];
  const float* Wr = (const float*)d_in[1];
  const float* br = (const float*)d_in[2];
  const float* W1 = (const float*)d_in[3];
  const float* b1 = (const float*)d_in[4];
  const float* W2 = (const float*)d_in[5];
  const float* b2 = (const float*)d_in[6];
  float* out = (float*)d_out;

  // workspace layout (256B-aligned offsets); total ~245.7 MB
  char* w = (char*)d_ws;
  int*      counts     = (int*)(w + 0);
  int*      poffsets   = (int*)(w + 256);
  int*      expert_a   = (int*)(w + 512);
  int*      slot_a     = (int*)(w + 512 + 65536);
  float*    weight_a   = (float*)(w + 512 + 2 * 65536);
  int*      assign_row = (int*)(w + 512 + 3 * 65536);
  int*      row_token  = (int*)(w + 512 + 4 * 65536);          // ends 332288
  uint16_t* Xg   = (uint16_t*)(w + 332288);                    // 35,651,584 B
  uint16_t* W1bf = (uint16_t*)(w + 35983872ull);               // 33,554,432 B
  uint16_t* W2bf = (uint16_t*)(w + 69538304ull);               // 33,554,432 B
  uint16_t* Hbuf = (uint16_t*)(w + 103092736ull);              // 71,303,168 B
  float*    Yg   = (float*)   (w + 174395904ull);              // 71,303,168 B

  k_init<<<(ROWS_CAP + 255) / 256, 256, 0, stream>>>(counts, row_token);
  k_router<<<NTOK / 256, 256, 0, stream>>>(x, Wr, br, counts,
                                           expert_a, slot_a, weight_a);
  k_scan<<<1, 32, 0, stream>>>(counts, poffsets);
  k_scatter<<<NASSIGN / 256, 256, 0, stream>>>(expert_a, slot_a, poffsets,
                                               row_token, assign_row);
  k_gather_x<<<(ROWS_CAP * (HIDDEN / 8)) / 256, 256, 0, stream>>>(x, row_token, Xg);
  // W1 [8][1024][2048] -> W1bf [8][2048][1024]
  k_transpose_w<<<dim3(INNER / 32, HIDDEN / 32, NEXP), 256, 0, stream>>>(
      W1, W1bf, HIDDEN, INNER);
  // W2 [8][2048][1024] -> W2bf [8][1024][2048]
  k_transpose_w<<<dim3(HIDDEN / 32, INNER / 32, NEXP), 256, 0, stream>>>(
      W2, W2bf, INNER, HIDDEN);

  // pass A: H = relu(Xg @ W1 + b1)  (bf16)
  k_moe_gemm<HIDDEN, INNER, true>
      <<<dim3(INNER / 128, NTILES_M), 256, 0, stream>>>(
          Xg, W1bf, b1, poffsets, Hbuf, nullptr);
  // pass B: Yg = H @ W2 + b2        (fp32)
  k_moe_gemm<INNER, HIDDEN, false>
      <<<dim3(HIDDEN / 128, NTILES_M), 256, 0, stream>>>(
          Hbuf, W2bf, b2, poffsets, nullptr, Yg);

  k_combine<<<(NTOK * (HIDDEN / 4)) / 256, 256, 0, stream>>>(
      Yg, assign_row, weight_a, out);
}
